// LSTMBackbone_55250459296312
// MI455X (gfx1250) — matrix operand order repaired and checked by reading, not compiled, not run on hardware
//
#include <hip/hip_runtime.h>

// ============================================================================
// LSTM backbone for MI455X (gfx1250): all GEMM work on v_wmma_f32_16x16x32_bf16
// (fp32 accumulate).
//  - Projections/FC: tiled WMMA GEMMs, K-loop unrolled x2 with ping-pong
//    fragment buffers (no rotation copies, partial s_wait_loadcnt overlap).
//  - Recurrence: batch rows are independent, so grid = 2 dirs x 2 half-batches
//    of 16 rows (one WMMA M-tile) with NO inter-WG communication. h lives in
//    LDS as the WMMA A operand; c stays in D-fragment-layout VGPRs so the gate
//    nonlinearity + state update are lane-local.
// ============================================================================

typedef __attribute__((ext_vector_type(16))) __bf16 v16bf;
typedef __attribute__((ext_vector_type(8)))  float  v8f;
typedef __attribute__((ext_vector_type(4)))  unsigned int u32x4;

union Frag { v16bf v; u32x4 q[2]; };

static __device__ inline __bf16 f32_to_bf16(float f) {
  unsigned u = __float_as_uint(f);
  unsigned r = (u + 0x7FFFu + ((u >> 16) & 1u)) >> 16;
  unsigned short s = (unsigned short)r;
  __bf16 out;
  *reinterpret_cast<unsigned short*>(&out) = s;
  return out;
}
static __device__ inline float bf16_to_f32(__bf16 h) {
  unsigned short s = *reinterpret_cast<const unsigned short*>(&h);
  return __uint_as_float(((unsigned)s) << 16);
}
static __device__ inline v8f v8f_zero() {
  v8f z = {0.f,0.f,0.f,0.f,0.f,0.f,0.f,0.f};
  return z;
}
static __device__ inline float sigmoidf_(float x) {
  return 1.0f / (1.0f + __expf(-x));
}

// A fragment (16x32 bf16, M x K), row-major source with leading dim ld.
// Lane l: row = row0 + (l&15); element e holds K = k0 + (e/8)*16 + (l>>4)*8 + e%8.
static __device__ inline v16bf load_a_frag(const __bf16* __restrict__ base,
                                           int row0, int k0, int ld, int lane) {
  const __bf16* p = base + (long long)(row0 + (lane & 15)) * ld + k0 + ((lane >> 4) << 3);
  Frag f;
  f.q[0] = *reinterpret_cast<const u32x4*>(p);
  f.q[1] = *reinterpret_cast<const u32x4*>(p + 16);
  return f.v;
}
// B fragment (32x16 bf16, K x N) sourced from W stored [N,K] row-major (ld=K):
// Lane l: col = n0 + (l&15); element e holds K = k0 + (e/8)*16 + (l>>4)*8 + e%8
// (same K order as the A fragment).
static __device__ inline v16bf load_b_frag(const __bf16* __restrict__ base,
                                           int n0, int k0, int ld, int lane) {
  const __bf16* p = base + (long long)(n0 + (lane & 15)) * ld + k0 + ((lane >> 4) << 3);
  Frag f;
  f.q[0] = *reinterpret_cast<const u32x4*>(p);
  f.q[1] = *reinterpret_cast<const u32x4*>(p + 16);
  return f.v;
}

// ---------------------------------------------------------------------------
// fp32 -> bf16 conversion helpers
// ---------------------------------------------------------------------------
__global__ void cvt_f32_bf16(const float* __restrict__ s, __bf16* __restrict__ d,
                             long long n) {
  long long i = (long long)blockIdx.x * blockDim.x + threadIdx.x;
  long long stride = (long long)gridDim.x * blockDim.x;
  for (; i < n; i += stride) d[i] = f32_to_bf16(s[i]);
}

// Convert rows of length ks into zero-padded rows of length kd (kd >= ks).
__global__ void cvt_pad_bf16(const float* __restrict__ s, __bf16* __restrict__ d,
                             int rows, int ks, int kd) {
  long long n = (long long)rows * kd;
  long long i = (long long)blockIdx.x * blockDim.x + threadIdx.x;
  long long stride = (long long)gridDim.x * blockDim.x;
  for (; i < n; i += stride) {
    int r = (int)(i / kd), k = (int)(i % kd);
    d[i] = (k < ks) ? f32_to_bf16(s[(long long)r * ks + k]) : f32_to_bf16(0.f);
  }
}

// ---------------------------------------------------------------------------
// WMMA GEMM:  out[z][m][n] = sum_k A[z][m][k] * W[z][n][k] + bias0[n] + bias1[n]
// 256 threads = 8 waves (2/SIMD -> deep register budget), WG tile 32M x 256N,
// wave tile 16M x 64N.  K must be a multiple of 64; the K loop is unrolled x2
// with ping-pong fragment buffers so no register copies are needed and loads
// for one half overlap the WMMAs of the other.
// epi==0: store bf16.  epi==1: LeakyReLU(0.01), store fp32.
// ---------------------------------------------------------------------------
__global__ __launch_bounds__(256)
void gemm_bf16_wmma(const __bf16* __restrict__ A, unsigned long long aStrideZ, int lda,
                    const __bf16* __restrict__ W, unsigned long long wStrideZ,
                    const float* __restrict__ bias0, const float* __restrict__ bias1,
                    int biasStrideZ,
                    void* __restrict__ out, unsigned long long outStrideZ, int ldout,
                    int K, int epi) {
  const int z = blockIdx.z;
  const __bf16* Az = A + (unsigned long long)z * aStrideZ;
  const __bf16* Wz = W + (unsigned long long)z * wStrideZ;
  const float* b0p = bias0 ? bias0 + (long long)z * biasStrideZ : nullptr;
  const float* b1p = bias1 ? bias1 + (long long)z * biasStrideZ : nullptr;

  const int lane = threadIdx.x & 31;
  const int w    = threadIdx.x >> 5;
  const int m0 = blockIdx.x * 32 + (w >> 2) * 16;
  const int n0 = blockIdx.y * 256 + (w & 3) * 64;

  v8f acc[4];
#pragma unroll
  for (int s = 0; s < 4; ++s) acc[s] = v8f_zero();

  // prologue: buffer 0 holds fragments for k = 0
  v16bf a0 = load_a_frag(Az, m0, 0, lda, lane);
  v16bf b0[4];
#pragma unroll
  for (int s = 0; s < 4; ++s) b0[s] = load_b_frag(Wz, n0 + s * 16, 0, K, lane);

  for (int k0 = 0; k0 < K; k0 += 64) {
    const int k1 = k0 + 32;                          // always < K (K % 64 == 0)
    const int k2 = (k0 + 64 < K) ? (k0 + 64) : 0;    // tail prefetch is harmless
    // prefetch buffer 1 (k1) while consuming buffer 0 (k0)
    v16bf a1 = load_a_frag(Az, m0, k1, lda, lane);
    v16bf b1[4];
#pragma unroll
    for (int s = 0; s < 4; ++s) b1[s] = load_b_frag(Wz, n0 + s * 16, k1, K, lane);
#pragma unroll
    for (int s = 0; s < 4; ++s)
      acc[s] = __builtin_amdgcn_wmma_f32_16x16x32_bf16(
          false, a0, false, b0[s], (short)0, acc[s], false, false);
    // prefetch buffer 0 (k2) while consuming buffer 1 (k1)
    a0 = load_a_frag(Az, m0, k2, lda, lane);
#pragma unroll
    for (int s = 0; s < 4; ++s) b0[s] = load_b_frag(Wz, n0 + s * 16, k2, K, lane);
#pragma unroll
    for (int s = 0; s < 4; ++s)
      acc[s] = __builtin_amdgcn_wmma_f32_16x16x32_bf16(
          false, a1, false, b1[s], (short)0, acc[s], false, false);
  }

  const int nl = lane & 15, hl = lane >> 4;
#pragma unroll
  for (int s = 0; s < 4; ++s) {
    const int n = n0 + s * 16 + nl;
    const float bias = (b0p ? b0p[n] : 0.f) + (b1p ? b1p[n] : 0.f);
#pragma unroll
    for (int r = 0; r < 8; ++r) {
      const int m = m0 + r + hl * 8;
      float v = acc[s][r] + bias;
      const unsigned long long idx =
          (unsigned long long)z * outStrideZ + (unsigned long long)m * ldout + n;
      if (epi == 1) {
        v = (v >= 0.f) ? v : 0.01f * v;
        reinterpret_cast<float*>(out)[idx] = v;
      } else {
        reinterpret_cast<__bf16*>(out)[idx] = f32_to_bf16(v);
      }
    }
  }
}

// ---------------------------------------------------------------------------
// LSTM recurrence. grid = (2 directions, 2 half-batches), 1024 threads = 32
// waves.  Batch rows are independent, so each WG owns 16 rows (one WMMA
// M-tile) and needs no communication with other WGs.
//   xg : [2][B*T][2048] bf16  (x@W_ih^T + b_ih + b_hh, gate order i,f,g,o)
//   whh: [2][2048][512] bf16
//   hs : [B*T][1024] bf16 out (column offset dir*512)
// Wave w owns hidden slice j0 = w*16: 4 accumulators (one per gate) and c in
// one v8f with the same (m,n) lane mapping -> lane-local cell update.
// Kept single-buffered on purpose: 8 waves/SIMD hide the grouped-load waits
// and per-wave VGPRs stay ~105 (no spill risk at full occupancy).
// ---------------------------------------------------------------------------
__global__ __launch_bounds__(1024)
void lstm_rec(const __bf16* __restrict__ xg, const __bf16* __restrict__ whh,
              __bf16* __restrict__ hs, int T) {
  const int dir = blockIdx.x;
  const int bh  = blockIdx.y;              // half-batch: rows bh*16 .. bh*16+15
  const __bf16* xg_d  = xg  + (unsigned long long)dir * (32ull * T) * 2048ull;
  const __bf16* whh_d = whh + (unsigned long long)dir * 2048ull * 512ull;

  // 520-element row stride: 1040B = 65*16B -> aligned 16B chunks, 4-bank/row
  // rotation -> conflict-free A-fragment reads.
  __shared__ __align__(16) __bf16 hbuf[16][520];
  {
    unsigned short* hz = reinterpret_cast<unsigned short*>(&hbuf[0][0]);
    for (int i = threadIdx.x; i < 16 * 520; i += blockDim.x) hz[i] = 0;
  }
  __syncthreads();

  const int lane = threadIdx.x & 31;
  const int w    = threadIdx.x >> 5;
  const int j0   = w * 16;                 // hidden slice owned by this wave
  const int nl   = lane & 15, hl = lane >> 4;

  v8f c = v8f_zero();

  for (int ti = 0; ti < T; ++ti) {
    const int t = dir ? (T - 1 - ti) : ti;

    v8f acc[4];
#pragma unroll
    for (int g = 0; g < 4; ++g) acc[g] = v8f_zero();

    // gates += h_{t-1} @ W_hh^T   (K = 512, 16 WMMA k-steps).  All 4 B
    // fragments issued before the WMMAs -> partial loadcnt waits.
    for (int kt = 0; kt < 16; ++kt) {
      const int k0 = kt * 32;
      const __bf16* pa = &hbuf[nl][k0 + hl * 8];
      Frag af;
      af.q[0] = *reinterpret_cast<const u32x4*>(pa);
      af.q[1] = *reinterpret_cast<const u32x4*>(pa + 16);
      v16bf b[4];
#pragma unroll
      for (int g = 0; g < 4; ++g)
        b[g] = load_b_frag(whh_d, g * 512 + j0, k0, 512, lane);
#pragma unroll
      for (int g = 0; g < 4; ++g)
        acc[g] = __builtin_amdgcn_wmma_f32_16x16x32_bf16(
            false, af.v, false, b[g], (short)0, acc[g], false, false);
    }

    // add xg, apply activations, update c/h -- all lane-local in D layout.
    v8f hst;
#pragma unroll
    for (int r = 0; r < 8; ++r) {
      const int bb = bh * 16 + r + hl * 8;                  // global batch row
      const __bf16* xr =
          xg_d + ((unsigned long long)bb * T + t) * 2048ull + j0 + nl;
      const float gi = sigmoidf_(acc[0][r] + bf16_to_f32(xr[0]));
      const float gf = sigmoidf_(acc[1][r] + bf16_to_f32(xr[512]));
      const float gg = tanhf   (acc[2][r] + bf16_to_f32(xr[1024]));
      const float go = sigmoidf_(acc[3][r] + bf16_to_f32(xr[1536]));
      const float cn = gf * c[r] + gi * gg;
      c[r] = cn;
      hst[r] = go * tanhf(cn);
    }

    __syncthreads();  // everyone finished reading h_{t-1} from LDS
#pragma unroll
    for (int r = 0; r < 8; ++r) {
      const int m  = r + hl * 8;                            // local row
      const int bb = bh * 16 + m;
      const int j  = j0 + nl;
      const __bf16 hb = f32_to_bf16(hst[r]);
      hbuf[m][j] = hb;
      hs[((unsigned long long)bb * T + t) * 1024ull + dir * 512 + j] = hb;
    }
    __syncthreads();  // h_t visible for next step
  }
}

// ---------------------------------------------------------------------------
// Host orchestration (graph-capture safe: kernels on `stream` only).
// ---------------------------------------------------------------------------
extern "C" void kernel_launch(void* const* d_in, const int* in_sizes, int n_in,
                              void* d_out, int out_size, void* d_ws, size_t ws_size,
                              hipStream_t stream) {
  (void)in_sizes; (void)n_in; (void)out_size; (void)ws_size;
  const float* x      = (const float*)d_in[0];
  const float* w_ih0  = (const float*)d_in[1];
  const float* w_hh0  = (const float*)d_in[2];
  const float* b_ih0  = (const float*)d_in[3];
  const float* b_hh0  = (const float*)d_in[4];
  const float* w_ih12 = (const float*)d_in[5];
  const float* w_hh12 = (const float*)d_in[6];
  const float* b_ih12 = (const float*)d_in[7];
  const float* b_hh12 = (const float*)d_in[8];
  const float* w_fc   = (const float*)d_in[9];
  const float* b_fc   = (const float*)d_in[10];

  const int B = 32, T = 512, Dpad = 64, H = 512, G = 4 * H /*2048*/, O = 2048;
  const int BT = B * T;  // 16384

  char* p = (char*)d_ws;
  auto alloc = [&](size_t bytes) -> char* {
    char* r = p;
    p += (bytes + 255) & ~(size_t)255;
    return r;
  };
  __bf16* x_bf     = (__bf16*)alloc((size_t)BT * Dpad * 2);          //   2 MB
  __bf16* wih0_bf  = (__bf16*)alloc((size_t)2 * G * Dpad * 2);       // 0.5 MB (62->64 pad)
  __bf16* whh0_bf  = (__bf16*)alloc((size_t)2 * G * H * 2);          //   4 MB
  __bf16* wih12_bf = (__bf16*)alloc((size_t)4 * G * 1024 * 2);       //  16 MB
  __bf16* whh12_bf = (__bf16*)alloc((size_t)4 * G * H * 2);          //   8 MB
  __bf16* wfc_bf   = (__bf16*)alloc((size_t)O * 1024 * 2);           //   4 MB
  __bf16* xg       = (__bf16*)alloc((size_t)2 * BT * G * 2);         // 128 MB
  __bf16* hs_a     = (__bf16*)alloc((size_t)BT * 1024 * 2);          //  32 MB
  __bf16* hs_b     = (__bf16*)alloc((size_t)BT * 1024 * 2);          //  32 MB

  const int CB = 256;
  auto cgrid = [&](long long n) {
    long long g = (n + CB - 1) / CB;
    return (unsigned)(g > 16384 ? 16384 : g);
  };

  // weight / input conversion to bf16 (+ K-padding 62 -> 64 with zeros)
  cvt_pad_bf16<<<cgrid((long long)BT * Dpad), CB, 0, stream>>>(x, x_bf, BT, 62, Dpad);
  cvt_pad_bf16<<<cgrid((long long)2 * G * Dpad), CB, 0, stream>>>(w_ih0, wih0_bf, 2 * G, 62, Dpad);
  cvt_f32_bf16<<<cgrid((long long)2 * G * H), CB, 0, stream>>>(w_hh0, whh0_bf, (long long)2 * G * H);
  cvt_f32_bf16<<<cgrid((long long)4 * G * 1024), CB, 0, stream>>>(w_ih12, wih12_bf, (long long)4 * G * 1024);
  cvt_f32_bf16<<<cgrid((long long)4 * G * H), CB, 0, stream>>>(w_hh12, whh12_bf, (long long)4 * G * H);
  cvt_f32_bf16<<<cgrid((long long)O * 1024), CB, 0, stream>>>(w_fc, wfc_bf, (long long)O * 1024);

  const dim3 gblk(256);
  // ---- layer 0 ----
  gemm_bf16_wmma<<<dim3(BT / 32, G / 256, 2), gblk, 0, stream>>>(
      x_bf, 0ull, Dpad,
      wih0_bf, (unsigned long long)G * Dpad,
      b_ih0, b_hh0, G,
      xg, (unsigned long long)BT * G, G, Dpad, /*epi=*/0);
  lstm_rec<<<dim3(2, 2), 1024, 0, stream>>>(xg, whh0_bf, hs_a, T);

  // ---- layers 1,2 ----
  for (int l = 0; l < 2; ++l) {
    const __bf16* src = (l == 0) ? hs_a : hs_b;
    __bf16*       dst = (l == 0) ? hs_b : hs_a;
    gemm_bf16_wmma<<<dim3(BT / 32, G / 256, 2), gblk, 0, stream>>>(
        src, 0ull, 1024,
        wih12_bf + (size_t)l * 2 * G * 1024, (unsigned long long)G * 1024,
        b_ih12 + (size_t)l * 2 * G, b_hh12 + (size_t)l * 2 * G, G,
        xg, (unsigned long long)BT * G, G, 1024, /*epi=*/0);
    lstm_rec<<<dim3(2, 2), 1024, 0, stream>>>(xg, whh12_bf + (size_t)l * 2 * G * H, dst, T);
  }

  // ---- FC + LeakyReLU (fp32 out) ----
  gemm_bf16_wmma<<<dim3(BT / 32, O / 256, 1), gblk, 0, stream>>>(
      hs_a, 0ull, 1024,
      wfc_bf, 0ull,
      b_fc, nullptr, 0,
      d_out, 0ull, O, 1024, /*epi=*/1);
}
